// MegaTransformerCausalSelfAttention_75711683494489
// MI455X (gfx1250) — compile-verified, matrix operand order checked by reading, not run on hardware
//
#include <hip/hip_runtime.h>
#include <hip/hip_bf16.h>
#include <math.h>

// Problem constants (from reference)
#define BATCH  2
#define TSEQ   2048
#define DMODEL 2048
#define NH     16
#define NG     4
#define DHEAD  128
#define KVDIM  (NG * DHEAD)   // 512
#define ROWS4  (BATCH * TSEQ) // 4096

typedef __attribute__((ext_vector_type(16))) __bf16 v16bf;
typedef __attribute__((ext_vector_type(8)))  float  v8f;
typedef __attribute__((ext_vector_type(4)))  unsigned int u32x4;
typedef __attribute__((ext_vector_type(4)))  int i32x4;
typedef __attribute__((ext_vector_type(8)))  int i32x8;

union FragU { v16bf v; uint4 q[2]; };

// ---------------------------------------------------------------------------
// Tensor Data Mover support (5-arg builtin on ROCm7.2/clang-22, 6-arg on
// clang-23; manual staging fallback otherwise). Confirmed emitting: round-2
// histogram had exactly 5 unclassified ops == 5 TDM call sites.
// ---------------------------------------------------------------------------
#if defined(__gfx1250__) && __has_builtin(__builtin_amdgcn_tensor_load_to_lds) && \
    __has_builtin(__builtin_amdgcn_s_wait_tensorcnt)
#define USE_TDM 1
#else
#define USE_TDM 0
#endif

__device__ __forceinline__ void tdm_wait() {
#if USE_TDM
  __builtin_amdgcn_s_wait_tensorcnt(0);
#endif
}

// 2-D tile DMA: global row-major (stride0 in 2-byte elements) -> LDS with
// per-row padding (D# codes: interval 0=2dw,1=4,...; amount N -> N+1 dwords).
__device__ __forceinline__ void tdm_load_2d(unsigned int lds_off, const unsigned short* gsrc,
                                            unsigned int tile0, unsigned int tile1,
                                            unsigned int tdim0, unsigned int tdim1,
                                            unsigned long long stride0,
                                            unsigned int pad_interval_code,
                                            unsigned int pad_amount_code) {
#if USE_TDM
  unsigned long long ga = (unsigned long long)(uintptr_t)gsrc;
  u32x4 g0;
  g0[0] = 1u;                                                     // count=1, user D#
  g0[1] = lds_off;                                                // lds_addr
  g0[2] = (unsigned int)ga;                                       // global_addr[31:0]
  g0[3] = (unsigned int)((ga >> 32) & 0x1FFFFFFull) | (2u << 30); // addr[56:32] | type=2
  i32x8 g1;
  g1[0] = (int)((1u << 16) |                                      // data_size = 2 bytes
                (1u << 20) |                                      // pad_enable
                (pad_interval_code << 22) | (pad_amount_code << 25));
  g1[1] = (int)((tdim0 & 0xFFFFu) << 16);                         // tensor_dim0[15:0]
  g1[2] = (int)((tdim0 >> 16) | ((tdim1 & 0xFFFFu) << 16));       // dim0 hi | dim1 lo
  g1[3] = (int)((tdim1 >> 16) | (tile0 << 16));                   // dim1 hi | tile_dim0
  g1[4] = (int)(tile1);                                           // tile_dim1 | tile_dim2=0
  g1[5] = (int)(unsigned int)(stride0 & 0xFFFFFFFFull);           // dim0_stride[31:0]
  g1[6] = (int)(unsigned int)((stride0 >> 32) & 0xFFFFull);       // dim0_stride[47:32]
  g1[7] = 0;
  i32x4 z4 = {0, 0, 0, 0};
#if __clang_major__ >= 23
  i32x8 z8 = {0, 0, 0, 0, 0, 0, 0, 0};
  __builtin_amdgcn_tensor_load_to_lds(g0, g1, z4, z4, z8, 0);
#else
  __builtin_amdgcn_tensor_load_to_lds(g0, g1, z4, z4, 0);
#endif
#endif
}

__device__ __forceinline__ unsigned short f32_to_bf16(float f) {
  union { float f; unsigned int u; } x; x.f = f;
  unsigned int r = x.u + 0x7FFFu + ((x.u >> 16) & 1u);  // RNE
  return (unsigned short)(r >> 16);
}
__device__ __forceinline__ float bf16_to_f32(unsigned short h) {
  union { unsigned int u; float f; } x; x.u = ((unsigned int)h) << 16; return x.f;
}

// 16x32 bf16 fragment, CDNA5 16-bit A/B WMMA layout; two b128 loads per lane.
// Requires ld % 8 == 0 and 16B-aligned base.
__device__ __forceinline__ v16bf load_frag(const unsigned short* p0, int row0, int ld) {
  int lane = threadIdx.x & 31;
  const unsigned short* p = p0 + (size_t)(row0 + (lane & 15)) * (size_t)ld + ((lane >> 4) << 3);
  FragU f;
  f.q[0] = *(const uint4*)(p);        // k 0..7   (+kbase)
  f.q[1] = *(const uint4*)(p + 16);   // k 16..23 (+kbase)
  return f.v;
}

__device__ __forceinline__ v8f wmma_bf16f32(v16bf a, v16bf b, v8f c) {
  return __builtin_amdgcn_wmma_f32_16x16x32_bf16(false, a, false, b, (short)0, c, false, false);
}

// ---------------------------------------------------------------------------
// fp32 -> bf16 elementwise conversion (activations)
// ---------------------------------------------------------------------------
__global__ __launch_bounds__(256) void cvt_f32_bf16(const float* __restrict__ in,
                                                    unsigned short* __restrict__ out, int n) {
  int i = blockIdx.x * 256 + threadIdx.x;
  if (i < n) out[i] = f32_to_bf16(in[i]);
}

// ---------------------------------------------------------------------------
// fp32 [K,N] -> bf16 [N,K] transpose-convert (weights), LDS 32x32 tile
// ---------------------------------------------------------------------------
__global__ __launch_bounds__(256) void transpose_cvt(const float* __restrict__ in,
                                                     unsigned short* __restrict__ out,
                                                     int K, int N) {
  __shared__ float tile[32][33];
  int tx = threadIdx.x & 31, ty = threadIdx.x >> 5;   // ty 0..7
  int n0 = blockIdx.x * 32, k0 = blockIdx.y * 32;
#pragma unroll
  for (int i = 0; i < 4; ++i)
    tile[ty + 8 * i][tx] = in[(size_t)(k0 + ty + 8 * i) * (size_t)N + n0 + tx];
  __syncthreads();
#pragma unroll
  for (int i = 0; i < 4; ++i)
    out[(size_t)(n0 + ty + 8 * i) * (size_t)K + k0 + tx] = f32_to_bf16(tile[tx][ty + 8 * i]);
}

// ---------------------------------------------------------------------------
// Tiled WMMA GEMM: C[M,N] = A[M,K] * Bt^T  with Bt = B^T stored [N,K].
// 256 threads = 8 waves (2m x 4n), tile 128x128, k-step 32, double-buffered,
// BOTH tiles staged by TDM (identical D# shape, padded stride 40).
// TROUT: write C transposed ([N,M], for V). F32OUT: f32 + bias (final proj).
// ---------------------------------------------------------------------------
template <bool F32OUT, bool TROUT>
__global__ __launch_bounds__(256) void gemm_wmma(const unsigned short* __restrict__ A,
                                                 const unsigned short* __restrict__ Bt,
                                                 unsigned short* __restrict__ Cb,
                                                 float* __restrict__ Cf,
                                                 const float* __restrict__ bias,
                                                 int M, int N, int K) {
  __shared__ __align__(16) unsigned short sA[2][128][40];   // [m][k]
  __shared__ __align__(16) unsigned short sB[2][128][40];   // [n][k]

  const int tid  = threadIdx.x;
  const int lane = tid & 31;
  const int w    = tid >> 5;
  const int wm   = w >> 2;        // 0..1
  const int wn   = w & 3;         // 0..3
  const int m0   = blockIdx.x * 128;
  const int n0   = blockIdx.y * 128;

  v8f acc[4][2];
  const v8f zero8 = {0.f, 0.f, 0.f, 0.f, 0.f, 0.f, 0.f, 0.f};
#pragma unroll
  for (int mt = 0; mt < 4; ++mt)
#pragma unroll
    for (int nt = 0; nt < 2; ++nt) acc[mt][nt] = zero8;

  auto stage = [&](int buf, int k0) {
#if USE_TDM
    if (w == 0) {
      tdm_load_2d((unsigned int)(uintptr_t)&sA[buf][0][0], A + (size_t)m0 * (size_t)K + k0,
                  32, 128, (unsigned)K, (unsigned)M, (unsigned long long)K,
                  /*16 dw*/3, /*4 dw*/3);
      tdm_load_2d((unsigned int)(uintptr_t)&sB[buf][0][0], Bt + (size_t)n0 * (size_t)K + k0,
                  32, 128, (unsigned)K, (unsigned)N, (unsigned long long)K,
                  /*16 dw*/3, /*4 dw*/3);
    }
#else
    const int r = tid >> 1, cb = (tid & 1) * 16;
    const unsigned short* ga = A + (size_t)(m0 + r) * (size_t)K + k0 + cb;
    const unsigned short* gb = Bt + (size_t)(n0 + r) * (size_t)K + k0 + cb;
    if (k0 + 32 < K) { __builtin_prefetch(ga + 32, 0, 0); __builtin_prefetch(gb + 32, 0, 0); }
#pragma unroll
    for (int i = 0; i < 2; ++i) {
      *(uint4*)&sA[buf][r][cb + 8 * i] = *(const uint4*)(ga + 8 * i);
      *(uint4*)&sB[buf][r][cb + 8 * i] = *(const uint4*)(gb + 8 * i);
    }
#endif
  };
  auto stageWait = [&]() {
#if USE_TDM
    if (w == 0) tdm_wait();
#endif
    __syncthreads();
  };

  const int nk = K >> 5;
  int buf = 0;
  stage(0, 0);
  stageWait();

  for (int kt = 0; kt < nk; ++kt) {
    if (kt + 1 < nk) stage(buf ^ 1, (kt + 1) * 32);
    v16bf af[4], bfr[2];
#pragma unroll
    for (int mt = 0; mt < 4; ++mt) af[mt] = load_frag(&sA[buf][0][0], wm * 64 + mt * 16, 40);
#pragma unroll
    for (int nt = 0; nt < 2; ++nt) bfr[nt] = load_frag(&sB[buf][0][0], wn * 32 + nt * 16, 40);
#pragma unroll
    for (int mt = 0; mt < 4; ++mt)
#pragma unroll
      for (int nt = 0; nt < 2; ++nt) acc[mt][nt] = wmma_bf16f32(af[mt], bfr[nt], acc[mt][nt]);
    if (kt + 1 < nk) stageWait();
    buf ^= 1;
  }

  // epilogue: C layout = lanes 0-15 rows r, lanes 16-31 rows r+8, col = lane&15
  const int half = lane >> 4, nl = lane & 15;
#pragma unroll
  for (int mt = 0; mt < 4; ++mt) {
#pragma unroll
    for (int nt = 0; nt < 2; ++nt) {
#pragma unroll
      for (int r = 0; r < 8; ++r) {
        int row = m0 + wm * 64 + mt * 16 + half * 8 + r;
        int col = n0 + wn * 32 + nt * 16 + nl;
        float v = acc[mt][nt][r];
        if (F32OUT)     Cf[(size_t)row * (size_t)N + col] = v + bias[col];
        else if (TROUT) Cb[(size_t)col * (size_t)M + row] = f32_to_bf16(v);
        else            Cb[(size_t)row * (size_t)N + col] = f32_to_bf16(v);
      }
    }
  }
}

// ---------------------------------------------------------------------------
// RoPE (interleaved pairs, full head dim), in-place on bf16 [rows, nh*128]
// ---------------------------------------------------------------------------
__global__ __launch_bounds__(256) void rope_kernel(unsigned short* __restrict__ X,
                                                   int nh, int rowstride, int n) {
  int idx = blockIdx.x * 256 + threadIdx.x;
  if (idx >= n) return;
  int i   = idx & 63;               // pair index 0..63
  int h   = (idx >> 6) % nh;
  int row = idx / (64 * nh);
  int t   = row % TSEQ;
  float freq = (float)t * __expf(-(float)(2 * i) * (9.210340371976184f / 128.0f));
  float c = cosf(freq), s = sinf(freq);
  size_t off = (size_t)row * (size_t)rowstride + (size_t)h * DHEAD + 2 * i;
  float x0 = bf16_to_f32(X[off]);
  float x1 = bf16_to_f32(X[off + 1]);
  X[off]     = f32_to_bf16(x0 * c - x1 * s);
  X[off + 1] = f32_to_bf16(x1 * c + x0 * s);
}

// ---------------------------------------------------------------------------
// Flash attention with ALiBi + soft-cap + causal mask (GQA, REP=4).
// 128 threads = 4 waves; block = one (b,h,64-query tile); wave = 16-row strip.
// K [64][128] and Vt [128][64] tiles staged by TDM, double-buffered.
// ---------------------------------------------------------------------------
__global__ __launch_bounds__(128) void attn_kernel(const unsigned short* __restrict__ Q,
                                                   const unsigned short* __restrict__ Kb,
                                                   const unsigned short* __restrict__ Vt,
                                                   unsigned short* __restrict__ ctx) {
  __shared__ __align__(16) unsigned short sK[2][64][136];   // [key][dq]  (B-frag for S)
  __shared__ __align__(16) unsigned short sV[2][128][72];   // [dv][key]  (B-frag for PV)
  __shared__ __align__(16) unsigned short sP[4][16][72];    // per-wave P (A-frag for PV)

  const int tid  = threadIdx.x;
  const int lane = tid & 31;
  const int w    = tid >> 5;           // 0..3
  const int qt   = blockIdx.x;         // query tile (64 rows)
  const int bh   = blockIdx.y;
  const int b    = bh >> 4;
  const int h    = bh & 15;
  const int g    = h >> 2;             // GQA group (REP = 4)
  const int q0   = qt * 64;

  const float scale = 0.08838834764831845f;          // 1/sqrt(128)
  const float slope = exp2f(-0.5f * (float)(h + 1)); // alibi slope 2^(-(h+1)*8/16)

  // Preload this wave's Q strip as 4 A-fragments (k chunks of 32)
  v16bf qf[4];
  {
    const unsigned short* qrow = Q + (size_t)(b * TSEQ + q0 + w * 16) * DMODEL + h * DHEAD;
    __builtin_prefetch(qrow, 0, 0);
#pragma unroll
    for (int kc = 0; kc < 4; ++kc) qf[kc] = load_frag(qrow + kc * 32, 0, DMODEL);
  }

  auto stage = [&](int buf, int j0) {
#if USE_TDM
    if (w == 0) {
      tdm_load_2d((unsigned int)(uintptr_t)&sK[buf][0][0],
                  Kb + (size_t)(b * TSEQ + j0) * KVDIM + g * DHEAD,
                  /*tile*/DHEAD, 64, /*tensor*/KVDIM, ROWS4, KVDIM,
                  /*64 dw*/5, /*4 dw*/3);
      tdm_load_2d((unsigned int)(uintptr_t)&sV[buf][0][0],
                  Vt + (size_t)(g * DHEAD) * ROWS4 + (b * TSEQ + j0),
                  /*tile*/64, DHEAD, /*tensor*/ROWS4, KVDIM, ROWS4,
                  /*32 dw*/4, /*4 dw*/3);
    }
#else
    {
      const int r = tid >> 1, cb = (tid & 1) * 64;
      const unsigned short* gk = Kb + (size_t)(b * TSEQ + j0 + r) * KVDIM + g * DHEAD + cb;
#pragma unroll
      for (int i = 0; i < 8; ++i)
        *(uint4*)&sK[buf][r][cb + 8 * i] = *(const uint4*)(gk + 8 * i);
      const unsigned short* gv = Vt + (size_t)(g * DHEAD + tid) * ROWS4 + b * TSEQ + j0;
#pragma unroll
      for (int i = 0; i < 8; ++i)
        *(uint4*)&sV[buf][tid][8 * i] = *(const uint4*)(gv + 8 * i);
    }
#endif
  };
  auto stageWait = [&]() {
#if USE_TDM
    if (w == 0) tdm_wait();
#endif
    __syncthreads();
  };

  const v8f zero8 = {0.f, 0.f, 0.f, 0.f, 0.f, 0.f, 0.f, 0.f};
  v8f o[8];
#pragma unroll
  for (int dt = 0; dt < 8; ++dt) o[dt] = zero8;
  float mrow[8], lrow[8];
#pragma unroll
  for (int r = 0; r < 8; ++r) { mrow[r] = -1e30f; lrow[r] = 0.f; }

  const int half = lane >> 4, nl = lane & 15;
  const float t_q = (float)(q0 + w * 16 + half * 8);   // + r added later

  const int nblocks = qt + 1;   // causal: keys 0 .. q0+63
  int buf = 0;
  stage(0, 0);
  stageWait();

  for (int jb = 0; jb < nblocks; ++jb) {
    const int j0 = jb * 64;
    if (jb + 1 < nblocks) stage(buf ^ 1, j0 + 64);

    // --- S = Q * K^T : 16x64 per wave, 4 n-tiles x 4 k-chunks ---
    v8f s[4];
#pragma unroll
    for (int nt = 0; nt < 4; ++nt) s[nt] = zero8;
#pragma unroll
    for (int nt = 0; nt < 4; ++nt)
#pragma unroll
      for (int kc = 0; kc < 4; ++kc)
        s[nt] = wmma_bf16f32(qf[kc], load_frag(&sK[buf][0][0] + kc * 32, nt * 16, 136), s[nt]);

    // --- alibi + scale + soft-cap + causal; online softmax ---
#pragma unroll
    for (int r = 0; r < 8; ++r) {
      float tq = t_q + (float)r;
      float vmax = -1e30f;
#pragma unroll
      for (int nt = 0; nt < 4; ++nt) {
        float tk = (float)(j0 + nt * 16 + nl);
        float x = (s[nt][r] - slope * fabsf(tq - tk)) * scale;   // alibi BEFORE scaling
        x = 30.0f * tanhf(x * (1.0f / 30.0f));                    // grok soft-cap
        if (tk > tq) x = -1e30f;                                  // causal mask
        s[nt][r] = x;
        vmax = fmaxf(vmax, x);
      }
#pragma unroll
      for (int mk = 1; mk <= 8; mk <<= 1) vmax = fmaxf(vmax, __shfl_xor(vmax, mk, 32));
      float mnew  = fmaxf(mrow[r], vmax);
      float alpha = __expf(mrow[r] - mnew);
      float rs = 0.f;
#pragma unroll
      for (int nt = 0; nt < 4; ++nt) {
        float p = __expf(s[nt][r] - mnew);
        s[nt][r] = p;
        rs += p;
      }
#pragma unroll
      for (int mk = 1; mk <= 8; mk <<= 1) rs += __shfl_xor(rs, mk, 32);
      lrow[r] = lrow[r] * alpha + rs;
      mrow[r] = mnew;
#pragma unroll
      for (int dt = 0; dt < 8; ++dt) o[dt][r] *= alpha;
      // stash P (C layout -> LDS, bf16) for the A-fragment reload
#pragma unroll
      for (int nt = 0; nt < 4; ++nt)
        sP[w][half * 8 + r][nt * 16 + nl] = f32_to_bf16(s[nt][r]);
    }

    // --- O += P * V : A-frags from per-wave P LDS, B-frags from V LDS ---
    v16bf pf0 = load_frag(&sP[w][0][0], 0, 72);
    v16bf pf1 = load_frag(&sP[w][0][0] + 32, 0, 72);
#pragma unroll
    for (int dt = 0; dt < 8; ++dt) {
      o[dt] = wmma_bf16f32(pf0, load_frag(&sV[buf][0][0], dt * 16, 72), o[dt]);
      o[dt] = wmma_bf16f32(pf1, load_frag(&sV[buf][0][0] + 32, dt * 16, 72), o[dt]);
    }

    if (jb + 1 < nblocks) stageWait();
    buf ^= 1;
  }

  // --- finalize: O /= l ; write ctx (bf16, [B*T, H*DV] layout) ---
#pragma unroll
  for (int r = 0; r < 8; ++r) {
    float inv = 1.0f / lrow[r];
    size_t row = (size_t)(b * TSEQ + q0 + w * 16 + half * 8 + r);
#pragma unroll
    for (int dt = 0; dt < 8; ++dt)
      ctx[row * DMODEL + h * DHEAD + dt * 16 + nl] = f32_to_bf16(o[dt][r] * inv);
  }
}

// ---------------------------------------------------------------------------
// Launch
// ---------------------------------------------------------------------------
extern "C" void kernel_launch(void* const* d_in, const int* in_sizes, int n_in,
                              void* d_out, int out_size, void* d_ws, size_t ws_size,
                              hipStream_t stream) {
  const float* hidden = (const float*)d_in[0];   // [B,T,D]
  const float* Wq     = (const float*)d_in[1];   // [D, H*DQ]
  const float* Wk     = (const float*)d_in[2];   // [D, G*DQ]
  const float* Wv     = (const float*)d_in[3];   // [D, G*DV]
  const float* Wo     = (const float*)d_in[4];   // [H*DV, D]
  const float* bo     = (const float*)d_in[5];   // [D]
  float* out          = (float*)d_out;           // [B,T,D] fp32

  char* ws = (char*)d_ws;
  size_t off = 0;
  auto alloc = [&](size_t bytes) { char* p = ws + off; off += (bytes + 255) & ~(size_t)255; return p; };

  unsigned short* Xb   = (unsigned short*)alloc((size_t)ROWS4 * DMODEL * 2);
  unsigned short* Wqt  = (unsigned short*)alloc((size_t)DMODEL * DMODEL * 2);  // [2048, D]
  unsigned short* Wkt  = (unsigned short*)alloc((size_t)DMODEL * KVDIM * 2);   // [512, D]
  unsigned short* Wvt  = (unsigned short*)alloc((size_t)DMODEL * KVDIM * 2);   // [512, D]
  unsigned short* Wot  = (unsigned short*)alloc((size_t)DMODEL * DMODEL * 2);  // [D, 2048]
  unsigned short* Qb   = (unsigned short*)alloc((size_t)ROWS4 * DMODEL * 2);   // [4096, 2048]
  unsigned short* Kbf  = (unsigned short*)alloc((size_t)ROWS4 * KVDIM * 2);    // [4096, 512]
  unsigned short* Vtb  = (unsigned short*)alloc((size_t)ROWS4 * KVDIM * 2);    // [512, 4096]
  unsigned short* Ctx  = (unsigned short*)alloc((size_t)ROWS4 * DMODEL * 2);   // [4096, 2048]

  // activations: elementwise convert; weights: transpose-convert to [N,K]
  cvt_f32_bf16<<<(ROWS4 * DMODEL + 255) / 256, 256, 0, stream>>>(hidden, Xb, ROWS4 * DMODEL);
  transpose_cvt<<<dim3(DMODEL / 32, DMODEL / 32), 256, 0, stream>>>(Wq, Wqt, DMODEL, DMODEL);
  transpose_cvt<<<dim3(KVDIM / 32,  DMODEL / 32), 256, 0, stream>>>(Wk, Wkt, DMODEL, KVDIM);
  transpose_cvt<<<dim3(KVDIM / 32,  DMODEL / 32), 256, 0, stream>>>(Wv, Wvt, DMODEL, KVDIM);
  transpose_cvt<<<dim3(DMODEL / 32, DMODEL / 32), 256, 0, stream>>>(Wo, Wot, DMODEL, DMODEL);

  // QKV projections (Q,K bf16 row-major; V written transposed [512, 4096])
  gemm_wmma<false, false><<<dim3(ROWS4 / 128, DMODEL / 128), 256, 0, stream>>>(
      Xb, Wqt, Qb, nullptr, nullptr, ROWS4, DMODEL, DMODEL);
  gemm_wmma<false, false><<<dim3(ROWS4 / 128, KVDIM / 128), 256, 0, stream>>>(
      Xb, Wkt, Kbf, nullptr, nullptr, ROWS4, KVDIM, DMODEL);
  gemm_wmma<false, true><<<dim3(ROWS4 / 128, KVDIM / 128), 256, 0, stream>>>(
      Xb, Wvt, Vtb, nullptr, nullptr, ROWS4, KVDIM, DMODEL);

  // RoPE on Q and K (in place)
  {
    int nq = ROWS4 * NH * 64;
    rope_kernel<<<(nq + 255) / 256, 256, 0, stream>>>(Qb, NH, DMODEL, nq);
    int nk = ROWS4 * NG * 64;
    rope_kernel<<<(nk + 255) / 256, 256, 0, stream>>>(Kbf, NG, KVDIM, nk);
  }

  // Attention: grid (query tiles, B*H)
  attn_kernel<<<dim3(TSEQ / 64, BATCH * NH), 128, 0, stream>>>(Qb, Kbf, Vtb, Ctx);

  // Output projection with bias (f32 out)
  gemm_wmma<true, false><<<dim3(ROWS4 / 128, DMODEL / 128), 256, 0, stream>>>(
      Ctx, Wot, nullptr, out, bo, ROWS4, DMODEL, DMODEL);
}
